// AttnBlock2D_59545426592472
// MI455X (gfx1250) — compile-verified, matrix operand order checked by reading, not run on hardware
//
#include <hip/hip_runtime.h>

#define B_SZ   4
#define CH     512     // C_IN == C_HID
#define N_PIX  4096    // H*W = 64*64

typedef __bf16 bf16_t;
typedef __attribute__((ext_vector_type(16))) __bf16 v16bf;
typedef __attribute__((ext_vector_type(8)))  __bf16 v8bf;
typedef __attribute__((ext_vector_type(8)))  float  v8f;
typedef __attribute__((ext_vector_type(4)))  unsigned int v4u;
typedef __attribute__((ext_vector_type(8)))  unsigned int v8u;

static __device__ __forceinline__ v8f wmma_bf16(v16bf a, v16bf b, v8f c) {
  // D = A(16x32) * B(32x16) + C, f32 accumulate
  return __builtin_amdgcn_wmma_f32_16x16x32_bf16(false, a, false, b, (short)0, c,
                                                 false, false);
}

// Load a 16x32 A fragment (or a B fragment from memory stored [col][k]).
// base points at (row0, k0); rows are `ld` bf16 apart.
// Lane l<16 -> row l, K chunks {0..7, 16..23}; lane l>=16 -> row l-16,
// K chunks {8..15, 24..31}  (CDNA5 ISA 7.12.2 16-bit A layout).
static __device__ __forceinline__ v16bf load_frag(const bf16_t* base, int ld) {
  const int lane = threadIdx.x & 31;
  const int rc   = lane & 15;
  const int half = lane >> 4;
  const bf16_t* p = base + (size_t)rc * ld + half * 8;
  v8bf lo = *(const v8bf*)(p);
  v8bf hi = *(const v8bf*)(p + 16);
  v16bf r;
#pragma unroll
  for (int i = 0; i < 8; ++i) { r[i] = lo[i]; r[i + 8] = hi[i]; }
  return r;
}

// --- CDNA5 async global->LDS copy (ASYNCcnt-tracked, bypasses VGPRs) -------
// lds_addr: LDS byte offset (low 32 bits of generic pointer), per-lane.
static __device__ __forceinline__ void async_copy_b128(unsigned lds_addr,
                                                       const void* gaddr) {
  asm volatile("global_load_async_to_lds_b128 %0, %1, off"
               :
               : "v"(lds_addr), "v"(gaddr)
               : "memory");
}
static __device__ __forceinline__ void wait_async0() {
  asm volatile("s_wait_asynccnt 0" ::: "memory");
}
static __device__ __forceinline__ unsigned lds_off(const void* p) {
  return (unsigned)(uintptr_t)p;  // LDS generic addr: offset lives in [31:0]
}

// --- CDNA5 Tensor Data Mover: one-shot 2D tile DMA global->LDS -------------
// D# per ISA ch.8: group0 = {count/lds_addr/global_addr/type}, group1 =
// {data_size, tensor dims, tile dims, dim0 stride}. 2-byte elements.
// Wave-level instruction (EXEC ignored) tracked by TENSORcnt; issue from one
// wave per WG and publish with s_wait_tensorcnt + barrier.
static __device__ __forceinline__ void tdm_load_2d_bf16(
    unsigned lds_addr, const void* gaddr, unsigned tensor_d0,
    unsigned tensor_d1, unsigned tile_d0, unsigned tile_d1,
    unsigned stride0_elems) {
  const unsigned long long ga = (unsigned long long)(uintptr_t)gaddr;
  v4u g0;
  g0[0] = 1u;                                 // [1:0] count=1 (valid user D#)
  g0[1] = lds_addr;                           // [63:32] LDS byte address
  g0[2] = (unsigned)ga;                       // [95:64] global addr lo
  g0[3] = (unsigned)(ga >> 32) | (2u << 30);  // [120:96] addr hi, [127:126]=2
  v8u g1;
  g1[0] = 1u << 16;                           // [17:16] data_size=1 -> 2 bytes
  g1[1] = (tensor_d0 & 0xFFFFu) << 16;        // tensor_dim0[15:0] -> [63:48]
  g1[2] = (tensor_d0 >> 16) | ((tensor_d1 & 0xFFFFu) << 16);  // [95:64]
  g1[3] = (tensor_d1 >> 16) | (tile_d0 << 16);                // tile_dim0
  g1[4] = tile_d1;                            // [143:128] tile_dim1 (0=1D)
  g1[5] = stride0_elems;                      // tensor_dim0_stride[31:0]
  g1[6] = 0u;                                 // stride0[47:32]=0, stride1=0
  g1[7] = 0u;
  asm volatile("tensor_load_to_lds %0, %1" ::"s"(g0), "s"(g1) : "memory");
}

// ---------------------------------------------------------------------------
// K0a: convert the four 512x512 f32 weight matrices to bf16 (row-major kept)
// ---------------------------------------------------------------------------
__global__ __launch_bounds__(256) void cvt_w_kernel(
    const float* __restrict__ Wk, const float* __restrict__ Wq,
    const float* __restrict__ Wv, const float* __restrict__ Wo,
    bf16_t* __restrict__ Wkb, bf16_t* __restrict__ Wqb,
    bf16_t* __restrict__ Wvb, bf16_t* __restrict__ Wob) {
  const int i = blockIdx.x * 256 + threadIdx.x;
  if (i < CH * CH) {
    Wkb[i] = (bf16_t)Wk[i];
    Wqb[i] = (bf16_t)Wq[i];
    Wvb[i] = (bf16_t)Wv[i];
    Wob[i] = (bf16_t)Wo[i];
  }
}

// ---------------------------------------------------------------------------
// K0b: inp [B][C][N] f32  ->  Xt [B][N][C] bf16   (tiled 32x32 transpose)
// ---------------------------------------------------------------------------
__global__ __launch_bounds__(256) void transpose_in_kernel(
    const float* __restrict__ inp, bf16_t* __restrict__ Xt) {
  __shared__ bf16_t tile[32][33];
  const int b  = blockIdx.z;
  const int cb = blockIdx.y * 32;
  const int nb = blockIdx.x * 32;
  const int tc = threadIdx.x & 31;
  const int tr = threadIdx.x >> 5;  // 0..7
#pragma unroll
  for (int k = 0; k < 4; ++k) {
    const int c = cb + tr + k * 8;
    tile[tr + k * 8][tc] = (bf16_t)inp[((size_t)b * CH + c) * N_PIX + nb + tc];
  }
  __syncthreads();
#pragma unroll
  for (int k = 0; k < 4; ++k) {
    const int n = nb + tr + k * 8;
    Xt[((size_t)b * N_PIX + n) * CH + cb + tc] = tile[tc][tr + k * 8];
  }
}

// ---------------------------------------------------------------------------
// K1: fused QKV projection.
//   out[o,n] = sum_c W[o,c] * Xt[n,c]  (+bias)
//   Kt,Qt stored [B][N][C] bf16 (Q pre-scaled by 1/sqrt(CH)); V stored [B][C][N].
// One wave per 16-pixel tile; X B-fragments stay resident, 3 GEMMs share them.
// ---------------------------------------------------------------------------
__global__ __launch_bounds__(256) void qkv_proj_kernel(
    const bf16_t* __restrict__ Xt,
    const bf16_t* __restrict__ Wkb, const bf16_t* __restrict__ Wqb,
    const bf16_t* __restrict__ Wvb,
    const float* __restrict__ bk, const float* __restrict__ bq,
    const float* __restrict__ bv,
    bf16_t* __restrict__ Kt, bf16_t* __restrict__ Qt, bf16_t* __restrict__ Vc) {
  const int b    = blockIdx.y;
  const int wave = threadIdx.x >> 5;
  const int lane = threadIdx.x & 31;
  const int n0   = (blockIdx.x * 8 + wave) * 16;

  const bf16_t* X = Xt + ((size_t)b * N_PIX + n0) * CH;
  v16bf bx[16];
#pragma unroll
  for (int c16 = 0; c16 < 16; ++c16) bx[c16] = load_frag(X + c16 * 32, CH);

  bf16_t* Ktb = Kt + (size_t)b * N_PIX * CH;
  bf16_t* Qtb = Qt + (size_t)b * N_PIX * CH;
  bf16_t* Vcb = Vc + (size_t)b * CH * N_PIX;

  const int nl = lane & 15, half = lane >> 4;
  const float scale = 0.04419417382415922f;  // 1/sqrt(512)

  for (int ot = 0; ot < 32; ++ot) {
    const int o0 = ot * 16;
    v8f ck = {0.f, 0.f, 0.f, 0.f, 0.f, 0.f, 0.f, 0.f};
    v8f cq = ck, cv = ck;
#pragma unroll
    for (int c16 = 0; c16 < 16; ++c16) {
      ck = wmma_bf16(load_frag(Wkb + (size_t)o0 * CH + c16 * 32, CH), bx[c16], ck);
      cq = wmma_bf16(load_frag(Wqb + (size_t)o0 * CH + c16 * 32, CH), bx[c16], cq);
      cv = wmma_bf16(load_frag(Wvb + (size_t)o0 * CH + c16 * 32, CH), bx[c16], cv);
    }
    v8bf vk, vq;
#pragma unroll
    for (int r = 0; r < 8; ++r) {
      const int o = o0 + half * 8 + r;
      vk[r] = (bf16_t)(ck[r] + bk[o]);
      vq[r] = (bf16_t)((cq[r] + bq[o]) * scale);
      Vcb[(size_t)o * N_PIX + n0 + nl] = (bf16_t)(cv[r] + bv[o]);
    }
    // D fragment: VGPR r, lanes<16 -> M=r, lanes>=16 -> M=r+8; N=lane&15.
    // Store transposed into pixel-major [N][C]: 8 contiguous bf16 per lane.
    *(v8bf*)(Ktb + (size_t)(n0 + nl) * CH + o0 + half * 8) = vk;
    *(v8bf*)(Qtb + (size_t)(n0 + nl) * CH + o0 + half * 8) = vq;
  }
}

// ---------------------------------------------------------------------------
// K2: flash attention, double-buffered LDS staging:
//   Q tile (contiguous 32KB) via per-lane global_load_async_to_lds_b128,
//   V tile (strided 2D [512c][32j]) via one TDM tensor_load_to_lds.
// WG = 128 threads (4 waves). Each wave owns a 16-row (i) block.
// Per 32-column j step: issue next step's copies into the other buffer,
// compute 64 WMMAs on the current one, then wait async+tensor cnt + barrier.
// ---------------------------------------------------------------------------
__global__ __launch_bounds__(128) void attn_kernel(
    const bf16_t* __restrict__ Kt, const bf16_t* __restrict__ Qt,
    const bf16_t* __restrict__ Vc, bf16_t* __restrict__ Oattn) {
  __shared__ bf16_t Qs[2][32 * CH];    // 2 x 32 KB, layout [j][c]
  __shared__ bf16_t Vs[2][CH * 32];    // 2 x 32 KB, layout [c][j]
  __shared__ bf16_t Ps[4][16 * 32];    // 4 KB per-wave P transpose scratch

  const int b    = blockIdx.y;
  const int wave = threadIdx.x >> 5;
  const int lane = threadIdx.x & 31;
  const int tid  = threadIdx.x;
  const int row0 = blockIdx.x * 64 + wave * 16;

  const bf16_t* Ktb = Kt + (size_t)b * N_PIX * CH;
  const bf16_t* Qtb = Qt + (size_t)b * N_PIX * CH;
  const bf16_t* Vcb = Vc + (size_t)b * CH * N_PIX;

  // Resident A fragments: this wave's 16 K-rows, all 512 channels.
  v16bf ak[16];
#pragma unroll
  for (int c16 = 0; c16 < 16; ++c16)
    ak[c16] = load_frag(Ktb + (size_t)row0 * CH + c16 * 32, CH);

  v8f acc[32];
  const v8f vzero = {0.f, 0.f, 0.f, 0.f, 0.f, 0.f, 0.f, 0.f};
#pragma unroll
  for (int t = 0; t < 32; ++t) acc[t] = vzero;
  float mrow[8], lrow[8];
#pragma unroll
  for (int r = 0; r < 8; ++r) { mrow[r] = -1e30f; lrow[r] = 0.f; }

  const int nl = lane & 15, half = lane >> 4;

  // Q tile: [32][CH] bf16 = 2048 x 16B contiguous chunks, per-lane async.
  auto issue_q_async = [&](int jb, int buf) {
    const char* qsrc = (const char*)(Qtb + (size_t)(jb * 32) * CH);
    const unsigned qdst = lds_off(&Qs[buf][0]);
#pragma unroll
    for (int k = 0; k < 16; ++k) {
      const int idx = tid + k * 128;  // 0..2047 sixteen-byte chunks
      async_copy_b128(qdst + idx * 16, qsrc + (size_t)idx * 16);
    }
  };
  // V tile: 2D tile 32(j, contiguous) x 512(c), row stride N_PIX elems. TDM.
  auto issue_v_tdm = [&](int jb, int buf) {
    tdm_load_2d_bf16(lds_off(&Vs[buf][0]), Vcb + jb * 32,
                     /*tensor_d0=*/N_PIX, /*tensor_d1=*/CH,
                     /*tile_d0=*/32, /*tile_d1=*/CH,
                     /*stride0=*/N_PIX);
  };

  issue_q_async(0, 0);
  if (wave == 0) issue_v_tdm(0, 0);
  wait_async0();
  if (wave == 0) __builtin_amdgcn_s_wait_tensorcnt(0);
  __syncthreads();

  for (int jb = 0; jb < 128; ++jb) {
    const int cur = jb & 1;
    if (jb + 1 < 128) {                    // overlap next staging with compute
      issue_q_async(jb + 1, cur ^ 1);
      if (wave == 0) issue_v_tdm(jb + 1, cur ^ 1);
    }

    const bf16_t* Qcur = Qs[cur];
    const bf16_t* Vcur = Vs[cur];

    // --- S = K_block x Q_tile : two 16x16 fragments (j 0..15, 16..31) ---
    v8f s0 = vzero, s1 = vzero;
#pragma unroll
    for (int c16 = 0; c16 < 16; ++c16) {
      v16bf bq0 = load_frag(Qcur + 0 * CH  + c16 * 32, CH);
      v16bf bq1 = load_frag(Qcur + 16 * CH + c16 * 32, CH);
      s0 = wmma_bf16(ak[c16], bq0, s0);
      s1 = wmma_bf16(ak[c16], bq1, s1);
    }

    // --- online softmax (row i lives in VGPR r, 16-lane half-groups) ---
    float alpha[8];
#pragma unroll
    for (int r = 0; r < 8; ++r) {
      float t = fmaxf(s0[r], s1[r]);
      t = fmaxf(t, __shfl_xor(t, 1, 16));
      t = fmaxf(t, __shfl_xor(t, 2, 16));
      t = fmaxf(t, __shfl_xor(t, 4, 16));
      t = fmaxf(t, __shfl_xor(t, 8, 16));
      const float mnew = fmaxf(mrow[r], t);
      const float a  = __expf(mrow[r] - mnew);
      const float p0 = __expf(s0[r] - mnew);
      const float p1 = __expf(s1[r] - mnew);
      float ps = p0 + p1;
      ps += __shfl_xor(ps, 1, 16);
      ps += __shfl_xor(ps, 2, 16);
      ps += __shfl_xor(ps, 4, 16);
      ps += __shfl_xor(ps, 8, 16);
      lrow[r] = lrow[r] * a + ps;
      mrow[r] = mnew;
      alpha[r] = a;
      s0[r] = p0;
      s1[r] = p1;
    }
#pragma unroll
    for (int t = 0; t < 32; ++t)
#pragma unroll
      for (int r = 0; r < 8; ++r) acc[t][r] *= alpha[r];

    // --- transpose P (D layout -> A layout) through per-wave LDS ---
#pragma unroll
    for (int r = 0; r < 8; ++r) {
      const int row = half * 8 + r;
      Ps[wave][row * 32 + nl]      = (bf16_t)s0[r];
      Ps[wave][row * 32 + nl + 16] = (bf16_t)s1[r];
    }
    v16bf pa = load_frag(Ps[wave], 32);  // wave-internal DS ordering

    // --- acc += P x V_tile ---
#pragma unroll
    for (int t = 0; t < 32; ++t) {
      v16bf bv = load_frag(Vcur + (size_t)(t * 16) * 32, 32);
      acc[t] = wmma_bf16(pa, bv, acc[t]);
    }

    wait_async0();                                     // Q tile landed
    if (wave == 0) __builtin_amdgcn_s_wait_tensorcnt(0);  // V tile landed
    __syncthreads();   // all waves done with current buffer + see new data
  }

  // --- epilogue: normalize and store bf16 pixel-major [N][C] ---
  bf16_t* Ob = Oattn + (size_t)b * N_PIX * CH;
#pragma unroll
  for (int r = 0; r < 8; ++r) {
    const float inv = 1.0f / lrow[r];
    const int i = row0 + half * 8 + r;
#pragma unroll
    for (int t = 0; t < 32; ++t)
      Ob[(size_t)i * CH + t * 16 + nl] = (bf16_t)(acc[t][r] * inv);
  }
}

// ---------------------------------------------------------------------------
// K3: output projection. out[b,o,n] = sum_c Wo[o,c]*Oattn[n,c] + bo[o], f32.
// ---------------------------------------------------------------------------
__global__ __launch_bounds__(256) void out_proj_kernel(
    const bf16_t* __restrict__ Oattn, const bf16_t* __restrict__ Wob,
    const float* __restrict__ bo, float* __restrict__ out) {
  const int b    = blockIdx.y;
  const int wave = threadIdx.x >> 5;
  const int lane = threadIdx.x & 31;
  const int n0   = (blockIdx.x * 8 + wave) * 16;

  const bf16_t* Ob = Oattn + ((size_t)b * N_PIX + n0) * CH;
  v16bf bx[16];
#pragma unroll
  for (int c16 = 0; c16 < 16; ++c16) bx[c16] = load_frag(Ob + c16 * 32, CH);

  const int nl = lane & 15, half = lane >> 4;
  for (int ot = 0; ot < 32; ++ot) {
    v8f acc = {0.f, 0.f, 0.f, 0.f, 0.f, 0.f, 0.f, 0.f};
#pragma unroll
    for (int c16 = 0; c16 < 16; ++c16)
      acc = wmma_bf16(load_frag(Wob + (size_t)(ot * 16) * CH + c16 * 32, CH),
                      bx[c16], acc);
#pragma unroll
    for (int r = 0; r < 8; ++r) {
      const int o = ot * 16 + half * 8 + r;
      out[((size_t)b * CH + o) * N_PIX + n0 + nl] = acc[r] + bo[o];
    }
  }
}

// ---------------------------------------------------------------------------
extern "C" void kernel_launch(void* const* d_in, const int* in_sizes, int n_in,
                              void* d_out, int out_size, void* d_ws,
                              size_t ws_size, hipStream_t stream) {
  (void)in_sizes; (void)n_in; (void)out_size; (void)ws_size;
  const float* inp = (const float*)d_in[0];
  const float* Wk  = (const float*)d_in[1];
  const float* bk  = (const float*)d_in[2];
  const float* Wq  = (const float*)d_in[3];
  const float* bq  = (const float*)d_in[4];
  const float* Wv  = (const float*)d_in[5];
  const float* bv  = (const float*)d_in[6];
  const float* Wo  = (const float*)d_in[7];
  const float* bo  = (const float*)d_in[8];
  float* out = (float*)d_out;

  char* ws = (char*)d_ws;
  const size_t WB  = (size_t)CH * CH * sizeof(bf16_t);           // 512 KB
  const size_t XB  = (size_t)B_SZ * N_PIX * CH * sizeof(bf16_t); // 16 MB
  bf16_t* Wkb   = (bf16_t*)(ws + 0 * WB);
  bf16_t* Wqb   = (bf16_t*)(ws + 1 * WB);
  bf16_t* Wvb   = (bf16_t*)(ws + 2 * WB);
  bf16_t* Wob   = (bf16_t*)(ws + 3 * WB);
  bf16_t* Xt    = (bf16_t*)(ws + 4 * WB);
  bf16_t* Kt    = (bf16_t*)(ws + 4 * WB + 1 * XB);
  bf16_t* Qt    = (bf16_t*)(ws + 4 * WB + 2 * XB);
  bf16_t* Vc    = (bf16_t*)(ws + 4 * WB + 3 * XB);
  bf16_t* Oattn = (bf16_t*)(ws + 4 * WB + 4 * XB);   // total ~82 MB

  cvt_w_kernel<<<dim3((CH * CH + 255) / 256), dim3(256), 0, stream>>>(
      Wk, Wq, Wv, Wo, Wkb, Wqb, Wvb, Wob);
  transpose_in_kernel<<<dim3(N_PIX / 32, CH / 32, B_SZ), dim3(256), 0, stream>>>(
      inp, Xt);
  qkv_proj_kernel<<<dim3(N_PIX / 128, B_SZ), dim3(256), 0, stream>>>(
      Xt, Wkb, Wqb, Wvb, bk, bq, bv, Kt, Qt, Vc);
  attn_kernel<<<dim3(N_PIX / 64, B_SZ), dim3(128), 0, stream>>>(
      Kt, Qt, Vc, Oattn);
  out_proj_kernel<<<dim3(N_PIX / 128, B_SZ), dim3(256), 0, stream>>>(
      Oattn, Wob, bo, out);
}
